// LevelFeatureAttn_49409303773805
// MI455X (gfx1250) — matrix operand order repaired and checked by reading, not compiled, not run on hardware
//
#include <hip/hip_runtime.h>

typedef long long i64;
typedef unsigned int u32;
typedef unsigned short u16;
typedef __attribute__((ext_vector_type(16))) __bf16 v16bf;
typedef __attribute__((ext_vector_type(8)))  float  v8f;
typedef __attribute__((ext_vector_type(8)))  u32    v8u;

#define DEV __device__ __forceinline__

DEV u32 pkbf(float lo, float hi) {
  // round-to-nearest-even fp32 -> bf16, pack two into one dword
  u32 a = __float_as_uint(lo), b = __float_as_uint(hi);
  a += 0x7FFFu + ((a >> 16) & 1u);
  b += 0x7FFFu + ((b >> 16) & 1u);
  return (a >> 16) | (b & 0xFFFF0000u);
}

DEV float siluf(float v) { return v / (1.f + __expf(-v)); }

// ---------------------------------------------------------------------------
// Strided fp32 -> packed bf16 [rows, K] row-major (pairs stored as dwords).
// src element (r,k) at src + (r/RN)*bs + (r%RN)*rs + (k/KN)*hs + (k%KN)*ks
// ---------------------------------------------------------------------------
__global__ __launch_bounds__(256) void pack_bf16(
    const float* __restrict__ src, u16* __restrict__ dst, i64 total, int K,
    int RN, i64 bs, i64 rs, int KN, i64 hs, i64 ks)
{
  i64 i2 = ((i64)blockIdx.x * blockDim.x + threadIdx.x) * 2;
  if (i2 >= total) return;
  i64 r = i2 / K;
  int k = (int)(i2 - r * K);
  const float* p = src + (i64)(r / RN) * bs + (i64)(r % RN) * rs;
  float x0 = p[(i64)(k / KN) * hs + (i64)(k % KN) * ks];
  float x1 = p[(i64)((k + 1) / KN) * hs + (i64)((k + 1) % KN) * ks];
  *(u32*)(dst + i2) = pkbf(x0, x1);
}

// ---------------------------------------------------------------------------
// Fast WMMA GEMM on packed bf16 operands.
// Abf: [rows, K] row-major bf16.  Wbf: [cols, K] row-major bf16 (i.e. B^T),
// with an optional per-row-block base offset w_bs * (rowbase/RN) (per-(b,h)).
// D store fully generalized (f32, or bf16 when store_bf=1):
//   (r/RN)*d_bs + (r%RN)*d_rs + (c/CN)*d_hs + (c%CN)*d_cs
// One wave computes a (16*TR) x (16*TC) tile. rows%(16*TR)==0, K%32==0.
// ---------------------------------------------------------------------------
template <int TR, int TC>
__global__ __launch_bounds__(256) void gemm_fast(
    const u16* __restrict__ Abf, const u16* __restrict__ Wbf,
    float* __restrict__ Df, u16* __restrict__ Dh,
    int rows, int cols, int K, int RN, i64 w_bs,
    i64 d_bs, i64 d_rs, int CN, i64 d_hs, i64 d_cs, int store_bf)
{
  const int lane = threadIdx.x & 31;
  const int wave = (int)((blockIdx.x * blockDim.x + threadIdx.x) >> 5);
  const int row_tiles = rows / (16 * TR);
  const int tr = wave % row_tiles;
  const int strip = wave / row_tiles;
  const int c0 = strip * (16 * TC);
  if (c0 >= cols) return;
  const int half = lane >> 4, lr = lane & 15;
  const int r0 = tr * 16 * TR;
  const u16* Wb = Wbf + (i64)(r0 / RN) * w_bs;

  v8f acc[TR][TC] = {};
  for (int kb = 0; kb < K; kb += 32) {
    v16bf af[TR];
#pragma unroll
    for (int ti = 0; ti < TR; ++ti) {
      // lane fragment = two runs of 8 consecutive bf16 -> two b128 loads
      i64 base = (i64)(r0 + ti * 16 + lr) * K + kb + half * 8;
      const uint4* pa = (const uint4*)(Abf + base);
      uint4 a0 = pa[0], a1 = pa[2];
      if (kb + 32 < K) __builtin_prefetch((const void*)(Abf + base + 32), 0, 1);
      v8u av;
      av[0] = a0.x; av[1] = a0.y; av[2] = a0.z; av[3] = a0.w;
      av[4] = a1.x; av[5] = a1.y; av[6] = a1.z; av[7] = a1.w;
      af[ti] = __builtin_bit_cast(v16bf, av);
    }
#pragma unroll
    for (int t = 0; t < TC; ++t) {
      // B fragment = two runs of 8 consecutive bf16 -> two b128 loads
      i64 base = (i64)(c0 + t * 16 + lr) * K + kb + half * 8;
      const uint4* pb = (const uint4*)(Wb + base);
      uint4 b0 = pb[0], b1 = pb[2];
      v8u bu;
      bu[0] = b0.x; bu[1] = b0.y; bu[2] = b0.z; bu[3] = b0.w;
      bu[4] = b1.x; bu[5] = b1.y; bu[6] = b1.z; bu[7] = b1.w;
      v16bf bf_ = __builtin_bit_cast(v16bf, bu);
#pragma unroll
      for (int ti = 0; ti < TR; ++ti)
        acc[ti][t] = __builtin_amdgcn_wmma_f32_16x16x32_bf16(
            false, af[ti], false, bf_, (short)0, acc[ti][t], false, false);
    }
  }
#pragma unroll
  for (int ti = 0; ti < TR; ++ti)
#pragma unroll
    for (int t = 0; t < TC; ++t)
#pragma unroll
      for (int vr = 0; vr < 8; ++vr) {
        int rr = r0 + ti * 16 + vr + half * 8;
        int cc = c0 + t * 16 + lr;
        i64 ad = (i64)(rr / RN) * d_bs + (i64)(rr % RN) * d_rs +
                 (i64)(cc / CN) * d_hs + (i64)(cc % CN) * d_cs;
        float v = acc[ti][t][vr];
        if (store_bf) Dh[ad] = (u16)(pkbf(v, 0.f) & 0xFFFFu);
        else Df[ad] = v;
      }
}

// ------------------------- elementwise / reduction kernels ------------------

__global__ void memzero_k(float* p, int n) {
  int i = blockIdx.x * blockDim.x + threadIdx.x;
  if (i < n) p[i] = 0.f;
}

// rows of 256: softmax(raw*scale + rel_bias[n,:]) in place; one wave per row
__global__ __launch_bounds__(256) void softmax_k(float* __restrict__ attn,
                                                 const float* __restrict__ rel,
                                                 float scale)
{
  int lane = threadIdx.x & 31;
  i64 wid = (i64)((blockIdx.x * (i64)blockDim.x + threadIdx.x) >> 5);
  int n = (int)(wid & 1023);
  float* row = attn + wid * 256;
  const float* bias = rel + (i64)n * 256;
  float v[8];
  float mx = -3.0e38f;
#pragma unroll
  for (int i = 0; i < 8; ++i) {
    int m = lane + i * 32;
    v[i] = row[m] * scale + bias[m];
    mx = fmaxf(mx, v[i]);
  }
  for (int off = 16; off > 0; off >>= 1) mx = fmaxf(mx, __shfl_xor(mx, off, 32));
  float s = 0.f;
#pragma unroll
  for (int i = 0; i < 8; ++i) { v[i] = __expf(v[i] - mx); s += v[i]; }
  for (int off = 16; off > 0; off >>= 1) s += __shfl_xor(s, off, 32);
  float inv = 1.f / s;
#pragma unroll
  for (int i = 0; i < 8; ++i) row[lane + i * 32] = v[i] * inv;
}

// per-group sum/sumsq accumulation (groups contiguous in memory)
__global__ __launch_bounds__(256) void stats_pass(const float* __restrict__ src,
                                                  float* __restrict__ st, i64 gelems)
{
  int g = blockIdx.y;
  const float* p = src + (i64)g * gelems;
  float s = 0.f, ss = 0.f;
  for (i64 i = (i64)blockIdx.x * blockDim.x + threadIdx.x; i < gelems;
       i += (i64)gridDim.x * blockDim.x) {
    float x = p[i];
    s += x; ss += x * x;
  }
  for (int off = 16; off > 0; off >>= 1) { s += __shfl_xor(s, off, 32); ss += __shfl_xor(ss, off, 32); }
  __shared__ float sh[16];
  int w = threadIdx.x >> 5;
  if ((threadIdx.x & 31) == 0) { sh[w] = s; sh[8 + w] = ss; }
  __syncthreads();
  if (threadIdx.x == 0) {
    float ts = 0.f, tss = 0.f;
    for (int i = 0; i < 8; ++i) { ts += sh[i]; tss += sh[8 + i]; }
    atomicAdd(&st[2 * g], ts);
    atomicAdd(&st[2 * g + 1], tss);
  }
}

__global__ void stats_final(float* st, int ngroups, float inv_ne) {
  int g = blockIdx.x * blockDim.x + threadIdx.x;
  if (g >= ngroups) return;
  float m = st[2 * g] * inv_ne;
  float var = st[2 * g + 1] * inv_ne - m * m;
  st[2 * g] = m;
  st[2 * g + 1] = rsqrtf(var + 1e-5f);
}

#define NMPIX 262144  // 1024*256 attention "image" per (b)

// 1x1 conv 8->24 on attn image (one batch)
__global__ __launch_bounds__(256) void dla_econv(const float* __restrict__ attn_b,
                                                 const float* __restrict__ ew,
                                                 float* __restrict__ a1)
{
  int pix = blockIdx.x * blockDim.x + threadIdx.x;
  float in[8];
#pragma unroll
  for (int i = 0; i < 8; ++i) in[i] = attn_b[(i64)i * NMPIX + pix];
#pragma unroll
  for (int o = 0; o < 24; ++o) {
    float s = 0.f;
#pragma unroll
    for (int i = 0; i < 8; ++i) s += ew[o * 8 + i] * in[i];
    a1[(i64)o * NMPIX + pix] = s;
  }
}

// depthwise 3x3, pad 1, input = (optional silu)(GN(src)) applied on the fly
__global__ __launch_bounds__(256) void dwconv3(
    const float* __restrict__ src, const float* __restrict__ st,
    const float* __restrict__ gamma, const float* __restrict__ beta,
    const float* __restrict__ w9, float* __restrict__ dst,
    int Ch, int H, int W, int gpb, i64 total, int act)
{
  i64 idx = (i64)blockIdx.x * blockDim.x + threadIdx.x;
  if (idx >= total) return;
  i64 hw = (i64)H * W;
  i64 bc = idx / hw;          // b*Ch + c
  int c = (int)(bc % Ch);
  int b = (int)(bc / Ch);
  i64 pix = idx - bc * hw;
  int yy = (int)(pix / W), xx = (int)(pix % W);
  int g = b * gpb + c / (Ch / gpb);
  float m = st[2 * g], rs = st[2 * g + 1], ga = gamma[c], be = beta[c];
  const float* pc = src + bc * hw;
  const float* wc = w9 + c * 9;
  float s = 0.f;
#pragma unroll
  for (int dy = 0; dy < 3; ++dy) {
    int y2 = yy + dy - 1;
    if ((unsigned)y2 >= (unsigned)H) continue;
#pragma unroll
    for (int dx = 0; dx < 3; ++dx) {
      int x2 = xx + dx - 1;
      if ((unsigned)x2 >= (unsigned)W) continue;
      float v = pc[(i64)y2 * W + x2];
      v = (v - m) * rs * ga + be;
      if (act) v = siluf(v);
      s += wc[dy * 3 + dx] * v;
    }
  }
  dst[idx] = s;
}

// silu(GN(a2)) then 1x1 conv 24->8, writing back into the attn image (one batch)
__global__ __launch_bounds__(256) void dla_rconv(
    const float* __restrict__ a2, const float* __restrict__ st,
    const float* __restrict__ g24, const float* __restrict__ b24,
    const float* __restrict__ rw, float* __restrict__ out8)
{
  int pix = blockIdx.x * blockDim.x + threadIdx.x;
  float vin[24];
#pragma unroll
  for (int c = 0; c < 24; ++c) {
    float v = a2[(i64)c * NMPIX + pix];
    int g = c >> 3;
    v = (v - st[2 * g]) * st[2 * g + 1] * g24[c] + b24[c];
    vin[c] = siluf(v);
  }
#pragma unroll
  for (int o = 0; o < 8; ++o) {
    float s = 0.f;
#pragma unroll
    for (int c = 0; c < 24; ++c) s += rw[o * 24 + c] * vin[c];
    out8[(i64)o * NMPIX + pix] = s;
  }
}

// generic GN apply (optionally silu)
__global__ __launch_bounds__(256) void gn_apply(
    const float* __restrict__ src, float* __restrict__ dst,
    const float* __restrict__ st, const float* __restrict__ gamma,
    const float* __restrict__ beta, int Ch, i64 Sp, int gpb, i64 total, int act)
{
  i64 idx = (i64)blockIdx.x * blockDim.x + threadIdx.x;
  if (idx >= total) return;
  i64 bc = idx / Sp;
  int c = (int)(bc % Ch);
  int b = (int)(bc / Ch);
  int g = b * gpb + c / (Ch / gpb);
  float v = (src[idx] - st[2 * g]) * st[2 * g + 1] * gamma[c] + beta[c];
  if (act) v = siluf(v);
  dst[idx] = v;
}

// x = q_in + silu(proj_out + b_proj), output channel-major [B,C,N]
__global__ __launch_bounds__(256) void x_fuse(const float* __restrict__ qin,
                                              const float* __restrict__ xo,
                                              const float* __restrict__ bproj,
                                              float* __restrict__ x)
{
  i64 idx = (i64)blockIdx.x * blockDim.x + threadIdx.x;  // B*256*1024
  int n = (int)(idx & 1023);
  int c = (int)((idx >> 10) & 255);
  int b = (int)(idx >> 18);
  float o = xo[((i64)(b * 1024 + n)) * 256 + c] + bproj[c];
  x[idx] = qin[idx] + siluf(o);
}

// y1 = GN0(y0) + silu(GN1(t1))
__global__ __launch_bounds__(256) void y1_fuse(
    const float* __restrict__ y0, const float* __restrict__ t1,
    const float* __restrict__ st0, const float* __restrict__ st1,
    const float* __restrict__ g0, const float* __restrict__ b0,
    const float* __restrict__ g1, const float* __restrict__ b1,
    float* __restrict__ y1)
{
  i64 idx = (i64)blockIdx.x * blockDim.x + threadIdx.x;  // B*1024*1024
  int c = (int)((idx >> 10) & 1023);
  int b = (int)(idx >> 20);
  float a = (y0[idx] - st0[2 * b]) * st0[2 * b + 1] * g0[c] + b0[c];
  int g = b * 4 + (c >> 8);
  float d = (t1[idx] - st1[2 * g]) * st1[2 * g + 1] * g1[c] + b1[c];
  y1[idx] = a + siluf(d);
}

// out = x + silu(GN(y2))
__global__ __launch_bounds__(256) void out_fuse(
    const float* __restrict__ x, const float* __restrict__ y2,
    const float* __restrict__ st, const float* __restrict__ g,
    const float* __restrict__ bb, float* __restrict__ out)
{
  i64 idx = (i64)blockIdx.x * blockDim.x + threadIdx.x;  // B*256*1024
  int c = (int)((idx >> 10) & 255);
  int b = (int)(idx >> 18);
  float v = (y2[idx] - st[2 * b]) * st[2 * b + 1] * g[c] + bb[c];
  out[idx] = x[idx] + siluf(v);
}

// ---------------------------------------------------------------------------

extern "C" void kernel_launch(void* const* d_in, const int* in_sizes, int n_in,
                              void* d_out, int out_size, void* d_ws, size_t ws_size,
                              hipStream_t stream)
{
  (void)in_sizes; (void)n_in; (void)out_size; (void)ws_size;
  const float* q_in   = (const float*)d_in[0];
  const float* kv_in  = (const float*)d_in[1];
  const float* w_q    = (const float*)d_in[2];
  const float* w_kv   = (const float*)d_in[3];
  const float* w_proj = (const float*)d_in[4];
  const float* b_proj = (const float*)d_in[5];
  const float* rel    = (const float*)d_in[6];
  const float* dla_ew = (const float*)d_in[7];
  const float* dla_eg = (const float*)d_in[8];
  const float* dla_eb = (const float*)d_in[9];
  const float* dla_dw = (const float*)d_in[10];
  const float* dla_dg = (const float*)d_in[11];
  const float* dla_db = (const float*)d_in[12];
  const float* dla_rw = (const float*)d_in[13];
  const float* dla_rg = (const float*)d_in[14];
  const float* dla_rb = (const float*)d_in[15];
  const float* n1_g   = (const float*)d_in[16];
  const float* n1_b   = (const float*)d_in[17];
  const float* f0_w   = (const float*)d_in[18];
  const float* f0_g   = (const float*)d_in[19];
  const float* f0_b   = (const float*)d_in[20];
  const float* f1_w   = (const float*)d_in[21];
  const float* f1_g   = (const float*)d_in[22];
  const float* f1_b   = (const float*)d_in[23];
  const float* f2_w   = (const float*)d_in[24];
  const float* f2_g   = (const float*)d_in[25];
  const float* f2_b   = (const float*)d_in[26];
  float* out = (float*)d_out;

  const int B = 16, C = 256, N = 1024, M = 256, Hh = 8, hd = 32, FH = 1024;
  const i64 NM = (i64)N * M;

  char* ws = (char*)d_ws;
  size_t off = 0;
  auto allocF = [&](size_t nf) -> float* {
    float* p = (float*)(ws + off);
    off += nf * sizeof(float);
    off = (off + 255) & ~(size_t)255;
    return p;
  };
  auto allocH = [&](size_t nh) -> u16* {
    u16* p = (u16*)(ws + off);
    off += nh * sizeof(u16);
    off = (off + 255) & ~(size_t)255;
    return p;
  };
  // fp32 working tensors
  float* attn = allocF((size_t)B * Hh * N * M);   // 134 MB
  float* pv   = allocF((size_t)B * Hh * N * hd);
  float* a1   = allocF((size_t)24 * NM);          // DLA hidden (per batch)
  float* a2   = allocF((size_t)24 * NM);
  float* xb   = allocF((size_t)B * C * N);        // residual x
  float* buf0 = allocF((size_t)B * C * N);        // xo -> y -> y2
  float* y0   = allocF((size_t)B * FH * N);
  float* t1   = allocF((size_t)B * FH * N);
  float* y1b  = allocF((size_t)B * FH * N);
  float* st   = allocF(1024);
  // packed bf16 operands
  u16* aq    = allocH((size_t)B * N * C);         // q_in as [B*N, C]
  u16* akv   = allocH((size_t)B * M * C);         // kv_in as [B*M, C]
  u16* qbf   = allocH((size_t)B * Hh * N * hd);   // q  [B,H,N,hd]
  u16* kbf   = allocH((size_t)B * Hh * M * hd);   // k  [B,H,M,hd]  (= QK's W^T)
  u16* vTbf  = allocH((size_t)B * Hh * hd * M);   // v^T [B,H,hd,M] (= PV's W^T)
  u16* atbf  = allocH((size_t)B * Hh * N * M);    // DLA'd attn as [B*H*N, M]
  u16* pvbf  = allocH((size_t)B * N * C);         // pv as [B*N, C]
  u16* ybf   = allocH((size_t)B * N * C);         // y as [B*N, C]
  u16* y1bf  = allocH((size_t)B * N * FH);        // y1 as [B*N, FH]
  u16* wqT   = allocH((size_t)C * C);
  u16* wkT   = allocH((size_t)C * C);
  u16* wvT   = allocH((size_t)C * C);
  u16* wpT   = allocH((size_t)C * C);
  u16* f0T   = allocH((size_t)FH * C);
  u16* f2T   = allocH((size_t)C * FH);

  float* stD1 = st + 0;    // 16 batches x 3 groups
  float* stD2 = st + 96;
  float* stD3 = st + 192;  // 16 x 1
  float* stX  = st + 224;  // 16 x 8
  float* stF0 = st + 480;  // 16 x 1
  float* stF1 = st + 512;  // 16 x 4
  float* stF2 = st + 640;  // 16 x 1

  memzero_k<<<4, 256, 0, stream>>>(st, 1024);

  auto pack = [&](const float* src, u16* dst, i64 rows, int K,
                  int RN, i64 bs, i64 rs, int KN, i64 hs, i64 ks) {
    i64 tot = rows * (i64)K;
    pack_bf16<<<(int)(tot / 512), 256, 0, stream>>>(src, dst, tot, K, RN, bs, rs, KN, hs, ks);
  };
  auto g24 = [&](const u16* A, const u16* W, float* Df, u16* Dh,
                 int rows, int cols, int K, int RN, i64 w_bs,
                 i64 d_bs, i64 d_rs, int CN, i64 d_hs, i64 d_cs, int sbf) {
    int waves = (rows / 32) * (cols / 64);
    gemm_fast<2, 4><<<waves / 8, 256, 0, stream>>>(A, W, Df, Dh, rows, cols, K, RN,
        w_bs, d_bs, d_rs, CN, d_hs, d_cs, sbf);
  };
  auto g22 = [&](const u16* A, const u16* W, float* Df, u16* Dh,
                 int rows, int cols, int K, int RN, i64 w_bs,
                 i64 d_bs, i64 d_rs, int CN, i64 d_hs, i64 d_cs, int sbf) {
    int waves = (rows / 32) * (cols / 32);
    gemm_fast<2, 2><<<waves / 8, 256, 0, stream>>>(A, W, Df, Dh, rows, cols, K, RN,
        w_bs, d_bs, d_rs, CN, d_hs, d_cs, sbf);
  };

  // ---- pack activations and weights to bf16 ----
  pack(q_in,  aq,  (i64)B * N, C, N, (i64)C * N, 1, C, 0, N);
  pack(kv_in, akv, (i64)B * M, C, M, (i64)C * M, 1, C, 0, M);
  pack(w_q,      wqT, C, C, C, 0, 1, C, 0, C);
  pack(w_kv,     wkT, C, C, C, 0, 1, C, 0, 2 * C);
  pack(w_kv + C, wvT, C, C, C, 0, 1, C, 0, 2 * C);
  pack(w_proj,   wpT, C, C, C, 0, 1, C, 0, C);
  pack(f0_w, f0T, FH, C, FH, 0, C, C, 0, 1);
  pack(f2_w, f2T, C, FH, C, 0, FH, FH, 0, 1);

  // ---- projections (store bf16 directly in WMMA-ready layouts) ----
  g24(aq, wqT, nullptr, qbf, B * N, C, C, N, 0,
      (i64)Hh * N * hd, hd, hd, (i64)N * hd, 1, 1);
  g24(akv, wkT, nullptr, kbf, B * M, C, C, M, 0,
      (i64)Hh * M * hd, hd, hd, (i64)M * hd, 1, 1);
  g24(akv, wvT, nullptr, vTbf, B * M, C, C, M, 0,
      (i64)Hh * hd * M, 1, hd, (i64)hd * M, M, 1);
  // QK^T per (b,h): q[N,32] x k^T -> attn raw f32 [B,H,N,M]
  g24(qbf, kbf, attn, nullptr, B * Hh * N, M, hd, N, (i64)M * hd,
      NM, M, M, 0, 1, 0);
  // softmax(scale*raw + rel_bias)
  softmax_k<<<(B * Hh * N) / 8, 256, 0, stream>>>(attn, rel, 1.f / sqrtf((float)hd));

  // ---- DLA conv stack on attention image, per batch ----
  for (int b = 0; b < B; ++b) {
    float* attn_b = attn + (i64)b * Hh * NM;
    dla_econv<<<(int)(NM / 256), 256, 0, stream>>>(attn_b, dla_ew, a1);
    stats_pass<<<dim3(128, 3), 256, 0, stream>>>(a1, stD1 + b * 6, (i64)8 * NM);
    stats_final<<<1, 128, 0, stream>>>(stD1 + b * 6, 3, 1.f / (8.f * (float)NM));
    dwconv3<<<(int)((24 * NM) / 256), 256, 0, stream>>>(
        a1, stD1 + b * 6, dla_eg, dla_eb, dla_dw, a2, 24, 1024, 256, 3, (i64)24 * NM, 1);
    stats_pass<<<dim3(128, 3), 256, 0, stream>>>(a2, stD2 + b * 6, (i64)8 * NM);
    stats_final<<<1, 128, 0, stream>>>(stD2 + b * 6, 3, 1.f / (8.f * (float)NM));
    dla_rconv<<<(int)(NM / 256), 256, 0, stream>>>(a2, stD2 + b * 6, dla_dg, dla_db,
                                                   dla_rw, attn_b);
    stats_pass<<<dim3(128, 1), 256, 0, stream>>>(attn_b, stD3 + b * 2, (i64)8 * NM);
    stats_final<<<1, 128, 0, stream>>>(stD3 + b * 2, 1, 1.f / (8.f * (float)NM));
    gn_apply<<<(int)((8 * NM) / 256), 256, 0, stream>>>(
        attn_b, attn_b, stD3 + b * 2, dla_rg, dla_rb, 8, NM, 1, (i64)8 * NM, 0);
  }

  // ---- PV + output projection ----
  pack(attn, atbf, (i64)B * Hh * N, M, B * Hh * N, 0, M, M, 0, 1);
  g22(atbf, vTbf, pv, nullptr, B * Hh * N, hd, M, N, (i64)hd * M,
      (i64)N * hd, hd, hd, 0, 1, 0);
  pack(pv, pvbf, (i64)B * N, C, N, (i64)Hh * N * hd, hd, hd, (i64)N * hd, 1);
  g24(pvbf, wpT, buf0, nullptr, B * N, C, C, N, 0,
      (i64)N * C, C, C, 0, 1, 0);
  // x = q_in + silu(xo + b_proj), channel-major
  x_fuse<<<(B * C * N) / 256, 256, 0, stream>>>(q_in, buf0, b_proj, xb);

  // ---- y = GN(x, groups=8) ----
  stats_pass<<<dim3(8, 128), 256, 0, stream>>>(xb, stX, (i64)32 * N);
  stats_final<<<1, 128, 0, stream>>>(stX, 128, 1.f / (32.f * (float)N));
  gn_apply<<<(B * C * N) / 256, 256, 0, stream>>>(xb, buf0, stX, n1_g, n1_b,
                                                  C, N, 8, (i64)B * C * N, 0);

  // ---- IFFN ----
  pack(buf0, ybf, (i64)B * N, C, N, (i64)C * N, 1, C, 0, N);
  g24(ybf, f0T, y0, nullptr, B * N, FH, C, N, 0,
      (i64)FH * N, 1, FH, 0, N, 0);
  stats_pass<<<dim3(64, 16), 256, 0, stream>>>(y0, stF0, (i64)FH * N);
  stats_final<<<1, 128, 0, stream>>>(stF0, 16, 1.f / ((float)FH * (float)N));
  dwconv3<<<(int)(((i64)B * FH * N) / 256), 256, 0, stream>>>(
      y0, stF0, f0_g, f0_b, f1_w, t1, FH, 32, 32, 1, (i64)B * FH * N, 0);
  stats_pass<<<dim3(16, 64), 256, 0, stream>>>(t1, stF1, (i64)(FH / 4) * N);
  stats_final<<<1, 128, 0, stream>>>(stF1, 64, 1.f / ((float)(FH / 4) * (float)N));
  y1_fuse<<<(int)(((i64)B * FH * N) / 256), 256, 0, stream>>>(
      y0, t1, stF0, stF1, f0_g, f0_b, f1_g, f1_b, y1b);
  pack(y1b, y1bf, (i64)B * N, FH, N, (i64)FH * N, 1, FH, 0, N);
  g24(y1bf, f2T, buf0, nullptr, B * N, C, FH, N, 0,
      (i64)C * N, 1, C, 0, N, 0);
  stats_pass<<<dim3(16, 16), 256, 0, stream>>>(buf0, stF2, (i64)C * N);
  stats_final<<<1, 128, 0, stream>>>(stF2, 16, 1.f / ((float)C * (float)N));
  // out = x + silu(GN(y2))
  out_fuse<<<(B * C * N) / 256, 256, 0, stream>>>(xb, buf0, stF2, f2_g, f2_b, out);
}
